// DistGAT_5128190952005
// MI455X (gfx1250) — matrix operand order repaired and checked by reading, not compiled, not run on hardware
//
#include <hip/hip_runtime.h>
#include <cstdint>
#include <cstddef>

// Problem constants (fixed by the reference)
#define N0c 100000
#define N1c 25000
#define N2c 5000
#define E1c 800000
#define E2c 160000
#define INc 1024
#define HIDc 128
#define Hc 4
#define Cc 19
#define NEG_SLOPE 0.2f

#define NPAD1 512   // H*HID, already multiple of 16
#define NPAD2 80    // H*C = 76 padded to 80

typedef __attribute__((ext_vector_type(16))) _Float16 v16h;
typedef __attribute__((ext_vector_type(8)))  float    v8f;

// ---------- order-preserving float <-> uint encoding (for atomicMax) ----------
__device__ __forceinline__ unsigned fenc(float f) {
  unsigned u = __float_as_uint(f);
  return (u & 0x80000000u) ? ~u : (u | 0x80000000u);
}
__device__ __forceinline__ float fdec(unsigned u) {
  unsigned b = (u & 0x80000000u) ? (u & 0x7FFFFFFFu) : ~u;
  return __uint_as_float(b);
}

// ---------- W[K,N] f32 -> f16 fragment-swizzled B ----------
// Layout: [(kt*NT + nt)][lane 0..31][16 halves] -> a wave loads one 32x16
// B-tile fragment as one contiguous 32B chunk per lane.
// half h of lane l maps to B[kt*32 + ((l&16)?8:0) + (h<8 ? h : h+8), nt*16 + (l&15)]
__global__ void k_swizzle_w(const float* __restrict__ W, _Float16* __restrict__ Bswz,
                            int K, int N, int Npad) {
  int idx = blockIdx.x * blockDim.x + threadIdx.x;
  int total = K * Npad;
  if (idx >= total) return;
  int h    = idx & 15;
  int lane = (idx >> 4) & 31;
  int tile = idx >> 9;             // kt*NT + nt
  int NT = Npad >> 4;
  int kt = tile / NT, nt = tile % NT;
  int k = kt * 32 + ((lane & 16) ? 8 : 0) + (h < 8 ? h : h + 8);
  int n = nt * 16 + (lane & 15);
  float v = (n < N) ? W[(size_t)k * N + n] : 0.0f;
  Bswz[idx] = (_Float16)v;
}

// ---------- A-fragment helper: load 16x32 f32 slice as f16 fragment ----------
__device__ __forceinline__ v16h load_a_frag(const float* __restrict__ ap) {
  float4 q0 = *(const float4*)(ap);
  float4 q1 = *(const float4*)(ap + 4);
  float4 q2 = *(const float4*)(ap + 16);
  float4 q3 = *(const float4*)(ap + 20);
  v16h a;
  a[0]=(_Float16)q0.x;  a[1]=(_Float16)q0.y;  a[2]=(_Float16)q0.z;  a[3]=(_Float16)q0.w;
  a[4]=(_Float16)q1.x;  a[5]=(_Float16)q1.y;  a[6]=(_Float16)q1.z;  a[7]=(_Float16)q1.w;
  a[8]=(_Float16)q2.x;  a[9]=(_Float16)q2.y;  a[10]=(_Float16)q2.z; a[11]=(_Float16)q2.w;
  a[12]=(_Float16)q3.x; a[13]=(_Float16)q3.y; a[14]=(_Float16)q3.z; a[15]=(_Float16)q3.w;
  return a;
}

// ---------- WMMA GEMM: C[M,Npad] = A[M,K](f32, row-major) * Bswz(f16) ----------
// Block = 256 threads = 8 waves arranged WM x WN over (M, N). Each wave owns a
// 16 x (16*NTILES) strip. B fragments are double-buffered (b0/b1) with an
// unroll-by-2 K loop so the next step's loads are in flight during the WMMA
// chain. KT must be even (K multiple of 64).
template <int NTILES, int WM, int WN>
__global__ __launch_bounds__(256) void k_gemm(const float* __restrict__ A,
                                              const _Float16* __restrict__ Bswz,
                                              float* __restrict__ C,
                                              int M, int K, int NT, int lda, int ldc) {
  const int lane = threadIdx.x & 31;
  const int wave = threadIdx.x >> 5;
  const int wm   = wave % WM;
  const int wn   = wave / WM;
  const int m0   = blockIdx.x * (16 * WM) + wm * 16;
  const int nt0  = (blockIdx.y * WN + wn) * NTILES;

  int rA = m0 + (lane & 15);
  if (rA > M - 1) rA = M - 1;                 // clamp tail reads; stores guarded
  const int koff = (lane & 16) ? 8 : 0;       // A-fragment K-half select

  v8f acc[NTILES];
  #pragma unroll
  for (int t = 0; t < NTILES; ++t)
    #pragma unroll
    for (int r = 0; r < 8; ++r) acc[t][r] = 0.0f;

  const float* arow = A + (size_t)rA * lda;
  const int KT = K >> 5;                       // even for K in {1024, 512}
  const size_t bstep = (size_t)NT * 32 * 16;   // halves per K-step
  const _Float16* bp = Bswz + ((size_t)nt0 * 32 + lane) * 16;

  v16h b0[NTILES], b1[NTILES];
  #pragma unroll
  for (int t = 0; t < NTILES; ++t)
    b0[t] = *(const v16h*)(bp + (size_t)t * 512);

  for (int kt = 0; kt < KT; kt += 2) {
    // ---- prefetch B for step kt+1 (always exists: KT even) ----
    {
      const _Float16* bn = bp + (size_t)(kt + 1) * bstep;
      #pragma unroll
      for (int t = 0; t < NTILES; ++t)
        b1[t] = *(const v16h*)(bn + (size_t)t * 512);
    }
    // ---- even step: A load + WMMA chain on b0 ----
    {
      v16h a = load_a_frag(arow + kt * 32 + koff);
      #pragma unroll
      for (int t = 0; t < NTILES; ++t)
        acc[t] = __builtin_amdgcn_wmma_f32_16x16x32_f16(
            false, a, false, b0[t], (short)0, acc[t], false, false);
    }
    // ---- prefetch B for step kt+2 (uniform guard) ----
    if (kt + 2 < KT) {
      const _Float16* bn = bp + (size_t)(kt + 2) * bstep;
      #pragma unroll
      for (int t = 0; t < NTILES; ++t)
        b0[t] = *(const v16h*)(bn + (size_t)t * 512);
    }
    // ---- odd step: A load + WMMA chain on b1 ----
    {
      v16h a = load_a_frag(arow + (kt + 1) * 32 + koff);
      #pragma unroll
      for (int t = 0; t < NTILES; ++t)
        acc[t] = __builtin_amdgcn_wmma_f32_16x16x32_f16(
            false, a, false, b1[t], (short)0, acc[t], false, false);
    }
  }

  // C/D layout: VGPR r, lanes 0-15 -> row r, lanes 16-31 -> row r+8; col = lane&15
  const int rbase = m0 + ((lane & 16) ? 8 : 0);
  if (m0 + 16 <= M) {
    // fast path: whole 16-row strip in bounds, straight-line stores
    #pragma unroll
    for (int t = 0; t < NTILES; ++t) {
      int col = (nt0 + t) * 16 + (lane & 15);
      float* cp = C + (size_t)rbase * ldc + col;
      #pragma unroll
      for (int r = 0; r < 8; ++r) cp[(size_t)r * ldc] = acc[t][r];
    }
  } else {
    #pragma unroll
    for (int t = 0; t < NTILES; ++t) {
      int col = (nt0 + t) * 16 + (lane & 15);
      #pragma unroll
      for (int r = 0; r < 8; ++r) {
        int row = rbase + r;
        if (row < M) C[(size_t)row * ldc + col] = acc[t][r];
      }
    }
  }
}

// ---------- per-node attention scores: el = <fs, al>, er = <fs, ar> ----------
__global__ void k_attn(const float* __restrict__ Cm, const float* __restrict__ al,
                       const float* __restrict__ ar, float* __restrict__ el,
                       float* __restrict__ er, int M, int F, int ldc) {
  int w    = (blockIdx.x * blockDim.x + threadIdx.x) >> 5;
  int lane = threadIdx.x & 31;
  if (w >= M) return;
  const float* row = Cm + (size_t)w * ldc;
  for (int h = 0; h < Hc; ++h) {
    float sl = 0.0f, sr = 0.0f;
    for (int f = lane; f < F; f += 32) {
      float v = row[h * F + f];
      sl += v * al[h * F + f];
      sr += v * ar[h * F + f];
    }
    for (int off = 16; off > 0; off >>= 1) {
      sl += __shfl_xor(sl, off, 32);
      sr += __shfl_xor(sr, off, 32);
    }
    if (lane == 0) { el[w * Hc + h] = sl; er[w * Hc + h] = sr; }
  }
}

// ---------- edge pass 1: segment max (encoded uint atomicMax) ----------
__global__ void k_edge_max(const int* __restrict__ src, const int* __restrict__ dst,
                           const float* __restrict__ el, const float* __restrict__ er,
                           unsigned* __restrict__ menc, int E) {
  int e = blockIdx.x * blockDim.x + threadIdx.x;
  if (e >= E) return;
  int s = src[e], d = dst[e];
  #pragma unroll
  for (int h = 0; h < Hc; ++h) {
    float v = el[s * Hc + h] + er[d * Hc + h];
    v = (v > 0.0f) ? v : NEG_SLOPE * v;
    atomicMax(&menc[d * Hc + h], fenc(v));
  }
}

// ---------- edge pass 2: ex = exp(e - m), denom += ex ----------
__global__ void k_edge_exp(const int* __restrict__ src, const int* __restrict__ dst,
                           const float* __restrict__ el, const float* __restrict__ er,
                           const unsigned* __restrict__ menc, float* __restrict__ exa,
                           float* __restrict__ denom, int E) {
  int e = blockIdx.x * blockDim.x + threadIdx.x;
  if (e >= E) return;
  int s = src[e], d = dst[e];
  #pragma unroll
  for (int h = 0; h < Hc; ++h) {
    float v = el[s * Hc + h] + er[d * Hc + h];
    v = (v > 0.0f) ? v : NEG_SLOPE * v;
    float ex = expf(v - fdec(menc[d * Hc + h]));
    exa[(size_t)e * Hc + h] = ex;
    atomicAdd(&denom[d * Hc + h], ex);
  }
}

// ---------- edge pass 3: alpha = ex / denom[dst] ----------
__global__ void k_edge_norm(const int* __restrict__ dst, float* __restrict__ exa,
                            const float* __restrict__ denom, int E) {
  int idx = blockIdx.x * blockDim.x + threadIdx.x;
  if (idx >= E * Hc) return;
  int e = idx >> 2, h = idx & 3;   // Hc == 4
  exa[idx] = exa[idx] / denom[dst[e] * Hc + h];
}

// ---------- CSR build ----------
__global__ void k_deg(const int* __restrict__ dst, int* __restrict__ deg, int E) {
  int e = blockIdx.x * blockDim.x + threadIdx.x;
  if (e >= E) return;
  atomicAdd(&deg[dst[e]], 1);
}

__global__ __launch_bounds__(1024) void k_scan(const int* __restrict__ deg,
                                               int* __restrict__ rowptr, int n) {
  __shared__ int buf[1024];
  __shared__ int carry;
  int t = threadIdx.x;
  if (t == 0) carry = 0;
  __syncthreads();
  for (int base = 0; base < n; base += 1024) {
    int i = base + t;
    int v = (i < n) ? deg[i] : 0;
    buf[t] = v;
    __syncthreads();
    for (int off = 1; off < 1024; off <<= 1) {
      int tmp = (t >= off) ? buf[t - off] : 0;
      __syncthreads();
      buf[t] += tmp;
      __syncthreads();
    }
    if (i < n) rowptr[i] = carry + buf[t] - v;   // exclusive
    __syncthreads();
    if (t == 1023) carry += buf[1023];
    __syncthreads();
  }
  if (t == 0) rowptr[n] = carry;
}

__global__ void k_fill(const int* __restrict__ dst, const int* __restrict__ rowptr,
                       int* __restrict__ cursor, int* __restrict__ eidx, int E) {
  int e = blockIdx.x * blockDim.x + threadIdx.x;
  if (e >= E) return;
  int d = dst[e];
  int p = atomicAdd(&cursor[d], 1);
  eidx[rowptr[d] + p] = e;
}

// ---------- aggregation: one block per destination node, no output atomics ----------
__global__ __launch_bounds__(128) void k_aggregate(const int* __restrict__ rowptr,
        const int* __restrict__ eidx, const int* __restrict__ src,
        const float* __restrict__ alpha, const float* __restrict__ fs,
        const float* __restrict__ bias, float* __restrict__ out,
        int ldfs, int Ftot, int Fhead, int ldout, int relu) {
  int i = blockIdx.x, t = threadIdx.x;
  float acc[4] = {0.f, 0.f, 0.f, 0.f};
  int hh[4];
  #pragma unroll
  for (int r = 0; r < 4; ++r) {
    int f = t + r * 128;
    hh[r] = (f < Ftot) ? (f / Fhead) : 0;
  }
  int beg = rowptr[i], end = rowptr[i + 1];
  for (int j = beg; j < end; ++j) {
    int e = eidx[j];
    int s = src[e];
    const float* fr = fs + (size_t)s * ldfs;
    const float* ae = alpha + (size_t)e * Hc;
    #pragma unroll
    for (int r = 0; r < 4; ++r) {
      int f = t + r * 128;
      if (f < Ftot) acc[r] += ae[hh[r]] * fr[f];
    }
  }
  #pragma unroll
  for (int r = 0; r < 4; ++r) {
    int f = t + r * 128;
    if (f < Ftot) {
      float v = acc[r] + bias[f];
      if (relu) v = v > 0.0f ? v : 0.0f;
      out[(size_t)i * ldout + f] = v;
    }
  }
}

// ---------- final mean over heads ----------
__global__ void k_mean(const float* __restrict__ h2, float* __restrict__ out, int n) {
  int idx = blockIdx.x * blockDim.x + threadIdx.x;
  if (idx >= n) return;
  int i = idx / Cc, c = idx % Cc;
  const float* r = h2 + (size_t)i * (Hc * Cc);
  out[idx] = 0.25f * (r[c] + r[Cc + c] + r[2 * Cc + c] + r[3 * Cc + c]);
}

extern "C" void kernel_launch(void* const* d_in, const int* in_sizes, int n_in,
                              void* d_out, int out_size, void* d_ws, size_t ws_size,
                              hipStream_t stream) {
  (void)in_sizes; (void)n_in; (void)out_size; (void)ws_size;

  const float* x    = (const float*)d_in[0];
  const int*   src1 = (const int*)d_in[1];
  const int*   dst1 = (const int*)d_in[2];
  const int*   src2 = (const int*)d_in[3];
  const int*   dst2 = (const int*)d_in[4];
  const float* W1   = (const float*)d_in[5];
  const float* al1  = (const float*)d_in[6];
  const float* ar1  = (const float*)d_in[7];
  const float* b1   = (const float*)d_in[8];
  const float* W2   = (const float*)d_in[9];
  const float* al2  = (const float*)d_in[10];
  const float* ar2  = (const float*)d_in[11];
  const float* b2   = (const float*)d_in[12];
  float* out = (float*)d_out;

  // ---- workspace carve (256B aligned slices) ----
  char* base = (char*)d_ws;
  size_t off = 0;
  auto carve = [&](size_t bytes) -> void* {
    void* p = base + off;
    off = (off + bytes + 255) & ~(size_t)255;
    return p;
  };
  float*     fs1    = (float*)carve((size_t)N0c * NPAD1 * 4);
  float*     el1    = (float*)carve((size_t)N0c * Hc * 4);
  float*     er1    = (float*)carve((size_t)N0c * Hc * 4);
  unsigned*  menc1  = (unsigned*)carve((size_t)N1c * Hc * 4);
  float*     den1   = (float*)carve((size_t)N1c * Hc * 4);
  float*     exa1   = (float*)carve((size_t)E1c * Hc * 4);
  int*       deg1   = (int*)carve((size_t)N1c * 4);
  int*       cur1   = (int*)carve((size_t)N1c * 4);
  int*       rp1    = (int*)carve((size_t)(N1c + 1) * 4);
  int*       eidx1  = (int*)carve((size_t)E1c * 4);
  float*     h1     = (float*)carve((size_t)N1c * NPAD1 * 4);
  _Float16*  w1s    = (_Float16*)carve((size_t)INc * NPAD1 * 2);
  _Float16*  w2s    = (_Float16*)carve((size_t)NPAD1 * NPAD2 * 2);
  float*     fs2    = (float*)carve((size_t)N1c * NPAD2 * 4);
  float*     el2    = (float*)carve((size_t)N1c * Hc * 4);
  float*     er2    = (float*)carve((size_t)N1c * Hc * 4);
  unsigned*  menc2  = (unsigned*)carve((size_t)N2c * Hc * 4);
  float*     den2   = (float*)carve((size_t)N2c * Hc * 4);
  float*     exa2   = (float*)carve((size_t)E2c * Hc * 4);
  int*       deg2   = (int*)carve((size_t)N2c * 4);
  int*       cur2   = (int*)carve((size_t)N2c * 4);
  int*       rp2    = (int*)carve((size_t)(N2c + 1) * 4);
  int*       eidx2  = (int*)carve((size_t)E2c * 4);
  float*     h2     = (float*)carve((size_t)N2c * (Hc * Cc) * 4);

  // ---- zero the accumulator state (graph-capture-safe) ----
  hipMemsetAsync(menc1, 0, (size_t)N1c * Hc * 4, stream);   // 0 encodes < any real float
  hipMemsetAsync(den1,  0, (size_t)N1c * Hc * 4, stream);
  hipMemsetAsync(deg1,  0, (size_t)N1c * 4, stream);
  hipMemsetAsync(cur1,  0, (size_t)N1c * 4, stream);
  hipMemsetAsync(menc2, 0, (size_t)N2c * Hc * 4, stream);
  hipMemsetAsync(den2,  0, (size_t)N2c * Hc * 4, stream);
  hipMemsetAsync(deg2,  0, (size_t)N2c * 4, stream);
  hipMemsetAsync(cur2,  0, (size_t)N2c * 4, stream);

  // ---- weight swizzle to WMMA fragment layout ----
  {
    int tot1 = INc * NPAD1;
    k_swizzle_w<<<(tot1 + 255) / 256, 256, 0, stream>>>(W1, w1s, INc, NPAD1, NPAD1);
    int tot2 = NPAD1 * NPAD2;
    k_swizzle_w<<<(tot2 + 255) / 256, 256, 0, stream>>>(W2, w2s, NPAD1, Hc * Cc, NPAD2);
  }

  // ================= layer 1 =================
  {
    // waves 2x4 over (M,N): BM=32, BN=512 (all of N) -> x streamed from HBM once
    dim3 grid((N0c + 31) / 32, 1);
    k_gemm<8, 2, 4><<<grid, 256, 0, stream>>>(x, w1s, fs1, N0c, INc, NPAD1 / 16, INc, NPAD1);
  }
  k_attn<<<(N0c + 7) / 8, 256, 0, stream>>>(fs1, al1, ar1, el1, er1, N0c, HIDc, NPAD1);

  k_deg<<<(E1c + 255) / 256, 256, 0, stream>>>(dst1, deg1, E1c);
  k_scan<<<1, 1024, 0, stream>>>(deg1, rp1, N1c);
  k_fill<<<(E1c + 255) / 256, 256, 0, stream>>>(dst1, rp1, cur1, eidx1, E1c);

  k_edge_max<<<(E1c + 255) / 256, 256, 0, stream>>>(src1, dst1, el1, er1, menc1, E1c);
  k_edge_exp<<<(E1c + 255) / 256, 256, 0, stream>>>(src1, dst1, el1, er1, menc1, exa1, den1, E1c);
  k_edge_norm<<<(E1c * Hc + 255) / 256, 256, 0, stream>>>(dst1, exa1, den1, E1c);

  k_aggregate<<<N1c, 128, 0, stream>>>(rp1, eidx1, src1, exa1, fs1, b1, h1,
                                       NPAD1, Hc * HIDc, HIDc, NPAD1, /*relu=*/1);

  // ================= layer 2 =================
  {
    // waves 8x1 over (M,N): BM=128, BN=80 (5 tiles)
    dim3 grid((N1c + 127) / 128, 1);
    k_gemm<5, 8, 1><<<grid, 256, 0, stream>>>(h1, w2s, fs2, N1c, NPAD1, NPAD2 / 16, NPAD1, NPAD2);
  }
  k_attn<<<(N1c + 7) / 8, 256, 0, stream>>>(fs2, al2, ar2, el2, er2, N1c, Cc, NPAD2);

  k_deg<<<(E2c + 255) / 256, 256, 0, stream>>>(dst2, deg2, E2c);
  k_scan<<<1, 1024, 0, stream>>>(deg2, rp2, N2c);
  k_fill<<<(E2c + 255) / 256, 256, 0, stream>>>(dst2, rp2, cur2, eidx2, E2c);

  k_edge_max<<<(E2c + 255) / 256, 256, 0, stream>>>(src2, dst2, el2, er2, menc2, E2c);
  k_edge_exp<<<(E2c + 255) / 256, 256, 0, stream>>>(src2, dst2, el2, er2, menc2, exa2, den2, E2c);
  k_edge_norm<<<(E2c * Hc + 255) / 256, 256, 0, stream>>>(dst2, exa2, den2, E2c);

  k_aggregate<<<N2c, 128, 0, stream>>>(rp2, eidx2, src2, exa2, fs2, b2, h2,
                                       NPAD2, Hc * Cc, Cc, Hc * Cc, /*relu=*/0);

  k_mean<<<(N2c * Cc + 255) / 256, 256, 0, stream>>>(h2, out, N2c * Cc);
}
